// GraphAttentionAggregation_72902774882528
// MI455X (gfx1250) — matrix operand order repaired and checked by reading, not compiled, not run on hardware
//
#include <hip/hip_runtime.h>

#define HID    128
#define NEDGE  800000
#define NNODE  50000

typedef __attribute__((ext_vector_type(16))) __bf16       v16bf;
typedef __attribute__((ext_vector_type(8)))  float        v8f;
typedef __attribute__((ext_vector_type(4)))  unsigned int u32x4;
typedef __attribute__((ext_vector_type(4)))  float        f32x4;

union BFrag { v16bf v; u32x4 q[2]; };

// ---------------------------------------------------------------------------
// Kernel 0: repack W1 (384x128 f32, row-major) into bf16 WMMA B-fragments.
// Fragment (kt,nt) covers K=[kt*32,kt*32+32) x N=[nt*16,nt*16+16).
// Lane L, element e: K = kt*32 + 16*(e>=8) + 8*(L>=16) + (e&7) ; N = nt*16 + (L&15).
// Stored contiguously: wfrag[frag*512 + lane*16 + e]  -> 32B load per lane.
// ---------------------------------------------------------------------------
__global__ void k_prep_w1(const float* __restrict__ W1, __bf16* __restrict__ wfrag) {
    int t = blockIdx.x * blockDim.x + threadIdx.x;
    if (t >= 96 * 512) return;
    int frag = t >> 9;
    int rem  = t & 511;
    int lane = rem >> 4;
    int e    = rem & 15;
    int kt = frag >> 3, nt = frag & 7;
    int K = kt * 32 + ((e >> 3) << 4) + ((lane >> 4) << 3) + (e & 7);
    int N = nt * 16 + (lane & 15);
    wfrag[t] = (__bf16)W1[K * HID + N];
}

// ---------------------------------------------------------------------------
// Kernel 1: per-edge attention logits via bf16 WMMA.
// Block = 128 threads = 4 waves; 16 edges per wave, 64 per block.
// ---------------------------------------------------------------------------
__global__ __launch_bounds__(128) void k_logits(
    const float* __restrict__ h,    const int* __restrict__ ei,
    const float* __restrict__ attr, const __bf16* __restrict__ wfrag,
    const float* __restrict__ b1,   const float* __restrict__ W2,
    const float* __restrict__ b2,   float* __restrict__ logit)
{
    __shared__ __align__(16) __bf16 sA[64 * 384];   // 48 KB: [edge][k], k = h_i|h_j|attr

    const int tid = threadIdx.x;
    const int e0  = blockIdx.x * 64;

    // Gather: 64 edges * 96 float4 chunks, strided over 128 threads.
    for (int idx = tid; idx < 64 * 96; idx += 128) {
        int e    = idx / 96;
        int part = idx - e * 96;
        int src  = part >> 5;          // 0:h_i  1:h_j  2:edge_attr
        int f4   = part & 31;
        int edge = e0 + e;
        const float* p;
        if (src == 0)      p = h    + (size_t)ei[edge] * HID;
        else if (src == 1) p = h    + (size_t)ei[NEDGE + edge] * HID;
        else               p = attr + (size_t)edge * HID;
        f32x4 v = *(const f32x4*)(p + f4 * 4);
        __bf16* dst = &sA[e * 384 + src * 128 + f4 * 4];
        dst[0] = (__bf16)v.x; dst[1] = (__bf16)v.y;
        dst[2] = (__bf16)v.z; dst[3] = (__bf16)v.w;
    }
    __syncthreads();

    const int wave = tid >> 5, lane = tid & 31;
    const int half = lane >> 4, mn = lane & 15;

    v8f acc[8] = {};                                   // 16x128 f32 tile
    const __bf16* arow = &sA[(wave * 16 + mn) * 384];  // A: M = lane&15 (both halves)

    for (int kt = 0; kt < 12; ++kt) {
        // A fragment (16x32 bf16): lane holds K runs [8h,8h+8) and [16+8h,16+8h+8)
        BFrag a;
        a.q[0] = *(const u32x4*)(arow + kt * 32 + half * 8);
        a.q[1] = *(const u32x4*)(arow + kt * 32 + 16 + half * 8);
        #pragma unroll
        for (int nt = 0; nt < 8; ++nt) {
            BFrag b;
            const u32x4* bp = (const u32x4*)(wfrag + ((kt * 8 + nt) * 512 + lane * 16));
            b.q[0] = bp[0];
            b.q[1] = bp[1];
            acc[nt] = __builtin_amdgcn_wmma_f32_16x16x32_bf16(
                false, a.v, false, b.v, (short)0, acc[nt], false, false);
        }
    }

    // Epilogue: pre = acc + b1 ; elu ; dot with W2 over the N dimension.
    // C layout: VGPR r, lane L -> M = r + 8*half, N = lane&15 (per n-tile).
    float s[8];
    #pragma unroll
    for (int r = 0; r < 8; ++r) s[r] = 0.0f;
    #pragma unroll
    for (int nt = 0; nt < 8; ++nt) {
        int n = nt * 16 + mn;
        float bb = b1[n], w2 = W2[n];
        #pragma unroll
        for (int r = 0; r < 8; ++r) {
            float x  = acc[nt][r] + bb;
            float el = x > 0.0f ? x : (__expf(x) - 1.0f);
            s[r] += el * w2;
        }
    }
    // Reduce over the 16 lanes of each half (the N dimension).
    #pragma unroll
    for (int r = 0; r < 8; ++r) {
        s[r] += __shfl_xor(s[r], 8, 16);
        s[r] += __shfl_xor(s[r], 4, 16);
        s[r] += __shfl_xor(s[r], 2, 16);
        s[r] += __shfl_xor(s[r], 1, 16);
    }
    if (mn == 0) {
        float bb2 = b2[0];
        #pragma unroll
        for (int r = 0; r < 8; ++r) {
            float a_ = s[r] + bb2;
            a_ = a_ > 0.0f ? a_ : 0.2f * a_;          // leaky_relu(0.2)
            logit[e0 + wave * 16 + half * 8 + r] = a_;
        }
    }
}

// ---------------------------------------------------------------------------
// Softmax reductions over all NEDGE logits.
// ---------------------------------------------------------------------------
__global__ void k_pmax(const float* __restrict__ logit, float* __restrict__ pmax, int n) {
    __shared__ float s[256];
    float m = -3.402823466e38f;
    for (int i = blockIdx.x * blockDim.x + threadIdx.x; i < n; i += blockDim.x * gridDim.x)
        m = fmaxf(m, logit[i]);
    s[threadIdx.x] = m; __syncthreads();
    for (int st = 128; st > 0; st >>= 1) {
        if (threadIdx.x < st) s[threadIdx.x] = fmaxf(s[threadIdx.x], s[threadIdx.x + st]);
        __syncthreads();
    }
    if (threadIdx.x == 0) pmax[blockIdx.x] = s[0];
}

__global__ void k_fmax(const float* __restrict__ pmax, float* __restrict__ gmax, int n) {
    __shared__ float s[256];
    float m = -3.402823466e38f;
    for (int i = threadIdx.x; i < n; i += 256) m = fmaxf(m, pmax[i]);
    s[threadIdx.x] = m; __syncthreads();
    for (int st = 128; st > 0; st >>= 1) {
        if (threadIdx.x < st) s[threadIdx.x] = fmaxf(s[threadIdx.x], s[threadIdx.x + st]);
        __syncthreads();
    }
    if (threadIdx.x == 0) *gmax = s[0];
}

__global__ void k_exp_psum(float* __restrict__ logit, const float* __restrict__ gmax,
                           float* __restrict__ psum, int n) {
    __shared__ float s[256];
    float m = *gmax, acc = 0.0f;
    for (int i = blockIdx.x * blockDim.x + threadIdx.x; i < n; i += blockDim.x * gridDim.x) {
        float ev = __expf(logit[i] - m);
        logit[i] = ev;                 // reuse buffer for exp values
        acc += ev;
    }
    s[threadIdx.x] = acc; __syncthreads();
    for (int st = 128; st > 0; st >>= 1) {
        if (threadIdx.x < st) s[threadIdx.x] += s[threadIdx.x + st];
        __syncthreads();
    }
    if (threadIdx.x == 0) psum[blockIdx.x] = s[0];
}

__global__ void k_fsum(const float* __restrict__ psum, float* __restrict__ gsum, int n) {
    __shared__ float s[256];
    float acc = 0.0f;
    for (int i = threadIdx.x; i < n; i += 256) acc += psum[i];
    s[threadIdx.x] = acc; __syncthreads();
    for (int st = 128; st > 0; st >>= 1) {
        if (threadIdx.x < st) s[threadIdx.x] += s[threadIdx.x + st];
        __syncthreads();
    }
    if (threadIdx.x == 0) *gsum = s[0];
}

// ---------------------------------------------------------------------------
// Output zero + scatter: out[row] += alpha * h[col]
// ---------------------------------------------------------------------------
__global__ void k_zero(float* __restrict__ out) {
    int t = blockIdx.x * 256 + threadIdx.x;      // 1.6M float4 exactly
    ((f32x4*)out)[t] = (f32x4){0.0f, 0.0f, 0.0f, 0.0f};
}

__global__ void k_scatter(const float* __restrict__ expv, const float* __restrict__ gsum,
                          const int* __restrict__ ei, const float* __restrict__ h,
                          float* __restrict__ out) {
    int t = blockIdx.x * 256 + threadIdx.x;      // NEDGE*32 threads
    int e = t >> 5;
    int q = t & 31;
    int r = ei[e];
    int c = ei[NEDGE + e];
    float alpha = expv[e] * (1.0f / gsum[0]);
    f32x4 v = *(const f32x4*)(h + (size_t)c * HID + q * 4);
    float* o = out + (size_t)r * HID + q * 4;
    atomicAdd(o + 0, alpha * v.x);
    atomicAdd(o + 1, alpha * v.y);
    atomicAdd(o + 2, alpha * v.z);
    atomicAdd(o + 3, alpha * v.w);
}

// ---------------------------------------------------------------------------
// Host launcher
// ---------------------------------------------------------------------------
extern "C" void kernel_launch(void* const* d_in, const int* in_sizes, int n_in,
                              void* d_out, int out_size, void* d_ws, size_t ws_size,
                              hipStream_t stream) {
    const float* h    = (const float*)d_in[0];
    const int*   ei   = (const int*)  d_in[1];
    const float* attr = (const float*)d_in[2];
    const float* W1   = (const float*)d_in[3];
    const float* b1   = (const float*)d_in[4];
    const float* W2   = (const float*)d_in[5];
    const float* b2   = (const float*)d_in[6];
    float* out = (float*)d_out;

    char* ws = (char*)d_ws;
    // ws layout:
    // [0, 98304)               : W1 bf16 fragments (96 * 512 * 2B)
    // [98304, 3298304)         : logits / exp values (NEDGE f32)
    // [3298304, 3300352)       : partial max (512 f32)
    // [3300352, 3302400)       : partial sum (512 f32)
    // [3302400, 3302408)       : gmax, gsum
    __bf16* wfrag = (__bf16*)(ws);
    float*  logit = (float*)(ws + 98304);
    float*  pmax  = (float*)(ws + 3298304);
    float*  psum  = (float*)(ws + 3300352);
    float*  gmax  = (float*)(ws + 3302400);
    float*  gsum  = (float*)(ws + 3302404);

    k_prep_w1<<<192, 256, 0, stream>>>(W1, wfrag);
    k_logits<<<NEDGE / 64, 128, 0, stream>>>(h, ei, attr, wfrag, b1, W2, b2, logit);
    k_zero<<<(NNODE * HID) / 1024, 256, 0, stream>>>(out);
    k_pmax<<<512, 256, 0, stream>>>(logit, pmax, NEDGE);
    k_fmax<<<1, 256, 0, stream>>>(pmax, gmax, 512);
    k_exp_psum<<<512, 256, 0, stream>>>(logit, gmax, psum, NEDGE);
    k_fsum<<<1, 256, 0, stream>>>(psum, gsum, 512);
    k_scatter<<<(NEDGE * 32) / 256, 256, 0, stream>>>(logit, gsum, ei, h, out);
}
